// GCN_39410619908620
// MI455X (gfx1250) — matrix operand order repaired and checked by reading, not compiled, not run on hardware
//
#include <hip/hip_runtime.h>

// ---------- types ----------
typedef __attribute__((ext_vector_type(16))) __bf16 v16bf;
typedef __attribute__((ext_vector_type(8)))  __bf16 v8bf;
typedef __attribute__((ext_vector_type(8)))  float  v8f;
typedef __attribute__((ext_vector_type(4)))  int    v4i;

union AFrag { v16bf v; v8bf h[2]; };

#if __has_builtin(__builtin_amdgcn_global_load_async_to_lds_b128)
#define GCN_ASYNC_LDS 1
#endif

// async-to-LDS builtin takes (global v4i*, local v4i*, imm offset, imm cpol).
// Generic->AS casts go through uintptr_t; low 32 bits of a generic LDS pointer
// are the LDS offset (ISA aperture rules).
#define GCN_AS1P(p) ((__attribute__((address_space(1))) v4i*)(unsigned long long)(uintptr_t)(p))
#define GCN_AS3P(p) ((__attribute__((address_space(3))) v4i*)(unsigned int)(uintptr_t)(p))

__device__ __forceinline__ __bf16 f2bf(float f) {
    unsigned int u = __builtin_bit_cast(unsigned int, f);
    u += 0x7FFFu + ((u >> 16) & 1u);              // round-to-nearest-even
    unsigned short s = (unsigned short)(u >> 16);
    return __builtin_bit_cast(__bf16, s);
}

// ---------- elementwise helpers ----------
__global__ void k_zero_f32(float* p, long long n) {
    long long i = (long long)blockIdx.x * blockDim.x + threadIdx.x;
    if (i < n) p[i] = 0.0f;
}

__global__ void k_deg(const int* __restrict__ dst, float* __restrict__ deg, int E) {
    int e = blockIdx.x * blockDim.x + threadIdx.x;
    if (e < E) unsafeAtomicAdd(&deg[dst[e]], 1.0f);
}

__global__ void k_dinv(float* d, int n) {
    int i = blockIdx.x * blockDim.x + threadIdx.x;
    if (i < n) d[i] = rsqrtf(d[i] + 1.0f);  // deg + self loop
}

__global__ void k_f32_to_bf16(const float* __restrict__ in, __bf16* __restrict__ out, long long n) {
    long long i = (long long)blockIdx.x * blockDim.x + threadIdx.x;
    if (i < n) out[i] = f2bf(in[i]);
}

// zero-padded conversion for Wout: [rows_in,K] f32 -> [rows_out,K] bf16
__global__ void k_conv_pad(const float* __restrict__ in, __bf16* __restrict__ out,
                           int rows_in, int rows_out, int K) {
    long long n = (long long)rows_out * K;
    long long i = (long long)blockIdx.x * blockDim.x + threadIdx.x;
    if (i >= n) return;
    int row = (int)(i / K);
    out[i] = (row < rows_in) ? f2bf(in[i]) : f2bf(0.0f);
}

// ---------- sparse aggregation: one wave32 per edge ----------
__global__ void k_aggregate(const int* __restrict__ src, const int* __restrict__ dst,
                            const float* __restrict__ dinv, const float* __restrict__ h,
                            float* __restrict__ agg, int E) {
    int wave = blockIdx.x * (blockDim.x >> 5) + (threadIdx.x >> 5);
    if (wave >= E) return;                     // uniform across the wave
    int lane = threadIdx.x & 31;
    int s = src[wave], d = dst[wave];
    float nm = dinv[s] * dinv[d];
    const float* hs = h + (long long)s * 256;
    float* ad = agg + (long long)d * 256;
#pragma unroll
    for (int i = 0; i < 8; ++i) {
        int c = lane + 32 * i;                 // coalesced
        unsafeAtomicAdd(&ad[c], nm * hs[c]);
    }
}

// ---------- combine: agg + dinv^2 * h_lin + bias (+relu), emit bf16 ----------
__global__ void k_combine(const float* __restrict__ agg, const float* __restrict__ hlin,
                          const float* __restrict__ dinv, const float* __restrict__ bias,
                          __bf16* __restrict__ outbf, long long n, int do_relu) {
    long long i = (long long)blockIdx.x * blockDim.x + threadIdx.x;
    if (i >= n) return;
    int row = (int)(i >> 8);                   // H == 256
    int c   = (int)(i & 255);
    float di = dinv[row];
    float v = agg[i] + di * di * hlin[i] + bias[c];
    if (do_relu) v = fmaxf(v, 0.0f);
    outbf[i] = f2bf(v);
}

// ---------- WMMA GEMM: C[M, ncols] (f32) = A[M,K](bf16) @ W[ncols_pad,K](bf16)^T ----------
// wave tile: 16 rows x (16*NSUB) cols ; block: 8 waves -> 128 rows x (16*NSUB) cols.
// B tile (shared by all 8 waves) staged once in LDS, async-to-LDS when available.
template <int NSUB>
__global__ __launch_bounds__(256) void k_gemm_bf16(
        const __bf16* __restrict__ A, const __bf16* __restrict__ W,
        float* __restrict__ C, const float* __restrict__ bias,
        int M, int K, int ncols, int ldc) {
    constexpr int ROWS = 16 * NSUB;            // output cols handled by this block
    __shared__ __bf16 sB[ROWS * 264];          // KP = K+8 <= 264 (16B pad per row)
    const int KP = K + 8;

    int tid  = threadIdx.x;
    int wave = tid >> 5;
    int lane = tid & 31;
    int n0 = blockIdx.y * ROWS;

    // ---- cooperative stage of W[n0 .. n0+ROWS) x K into LDS (16B chunks) ----
    {
        int kchunks = K >> 3;                  // 8 bf16 = 16B per chunk
        int total = ROWS * kchunks;
        for (int c = tid; c < total; c += 256) {
            int row = c / kchunks;
            int kc  = c - row * kchunks;
            const __bf16* g = W + (long long)(n0 + row) * K + kc * 8;
            __bf16* l = &sB[row * KP + kc * 8];
#ifdef GCN_ASYNC_LDS
            __builtin_amdgcn_global_load_async_to_lds_b128(GCN_AS1P(g), GCN_AS3P(l), 0, 0);
#else
            *(v8bf*)l = *(const v8bf*)g;
#endif
        }
#ifdef GCN_ASYNC_LDS
#if __has_builtin(__builtin_amdgcn_s_wait_asynccnt)
        __builtin_amdgcn_s_wait_asynccnt(0);
#else
        asm volatile("s_wait_asynccnt 0x0" ::: "memory");
#endif
#endif
    }
    __syncthreads();

    int m0 = (blockIdx.x * 8 + wave) * 16;
    if (m0 >= M) return;                       // wave-uniform: EXEC stays all-ones

    int l15 = lane & 15;
    int hi  = lane >> 4;

    int rowA = m0 + l15;
    if (rowA >= M) rowA = M - 1;               // clamp loads; stores guarded below
    const __bf16* arow = A + (long long)rowA * K;

    v8f acc[NSUB] = {};

    for (int k0 = 0; k0 < K; k0 += 32) {
        // A fragment (16x32 bf16 ISA layout): two 16B global loads per lane
        AFrag af;
        af.h[0] = *reinterpret_cast<const v8bf*>(arow + k0 + hi * 8);
        af.h[1] = *reinterpret_cast<const v8bf*>(arow + k0 + hi * 8 + 16);
        if (k0 + 32 < K) __builtin_prefetch(arow + k0 + 32, 0, 0);

#pragma unroll
        for (int s = 0; s < NSUB; ++s) {
            // B fragment from LDS: lane(0-15)=col s*16+l15 @K k0..+7,k0+16..+23; lanes 16-31 @ +8
            const __bf16* bp = &sB[(s * 16 + l15) * KP + k0 + hi * 8];
            AFrag bf;
            bf.h[0] = *reinterpret_cast<const v8bf*>(bp);
            bf.h[1] = *reinterpret_cast<const v8bf*>(bp + 16);
            acc[s] = __builtin_amdgcn_wmma_f32_16x16x32_bf16(
                false, af.v, false, bf.v, (short)0, acc[s], false, false);
        }
    }

    // store: lane l, VGPR r -> row m0 + (l>=16?8:0)+r, col n0+s*16+(l&15)
#pragma unroll
    for (int s = 0; s < NSUB; ++s) {
#pragma unroll
        for (int r = 0; r < 8; ++r) {
            int row = m0 + hi * 8 + r;
            int col = n0 + s * 16 + l15;
            if (row < M && col < ncols) {
                float v = acc[s][r];
                if (bias) v += bias[col];
                C[(long long)row * ldc + col] = v;
            }
        }
    }
}

// ---------- host launch ----------
extern "C" void kernel_launch(void* const* d_in, const int* in_sizes, int n_in,
                              void* d_out, int out_size, void* d_ws, size_t ws_size,
                              hipStream_t stream) {
    constexpr int F = 128, H = 256, C = 40, CPAD = 48;
    const float* x    = (const float*)d_in[0];
    const int*   ei   = (const int*)  d_in[1];
    const float* W1   = (const float*)d_in[2];
    const float* b1   = (const float*)d_in[3];
    const float* W2   = (const float*)d_in[4];
    const float* b2   = (const float*)d_in[5];
    const float* Wout = (const float*)d_in[6];
    const float* bout = (const float*)d_in[7];
    const int N = in_sizes[0] / F;
    const int E = in_sizes[1] / 2;
    const int* src = ei;
    const int* dst = ei + E;

    // workspace layout (256B aligned)
    unsigned char* w = (unsigned char*)d_ws;
    size_t off = 0;
    auto take = [&](size_t bytes) { void* p = w + off; off += (bytes + 255) & ~(size_t)255; return p; };
    float*  dinv   = (float*) take((size_t)N * 4);
    float*  hbuf   = (float*) take((size_t)N * H * 4);   // linear-layer output
    float*  aggbuf = (float*) take((size_t)N * H * 4);   // scatter accumulator
    __bf16* xb     = (__bf16*)take((size_t)N * H * 2);   // bf16 layer input
    __bf16* wb     = (__bf16*)take((size_t)H * H * 2);   // bf16 weights (reused)

    const long long NH = (long long)N * H;
    auto blks = [](long long n, int b) { return (unsigned)((n + b - 1) / b); };

    // --- degree / normalization ---
    k_zero_f32<<<blks(N, 256), 256, 0, stream>>>(dinv, N);
    k_deg<<<blks(E, 256), 256, 0, stream>>>(dst, dinv, E);
    k_dinv<<<blks(N, 256), 256, 0, stream>>>(dinv, N);

    // --- layer 1: h1 = x @ W1^T ---
    k_f32_to_bf16<<<blks((long long)N * F, 256), 256, 0, stream>>>(x, xb, (long long)N * F);
    k_f32_to_bf16<<<blks((long long)H * F, 256), 256, 0, stream>>>(W1, wb, (long long)H * F);
    {
        dim3 g(blks(N, 128), H / 64);
        k_gemm_bf16<4><<<g, 256, 0, stream>>>(xb, wb, hbuf, nullptr, N, F, H, H);
    }
    k_zero_f32<<<blks(NH, 256), 256, 0, stream>>>(aggbuf, NH);
    k_aggregate<<<blks(E, 8), 256, 0, stream>>>(src, dst, dinv, hbuf, aggbuf, E);
    k_combine<<<blks(NH, 256), 256, 0, stream>>>(aggbuf, hbuf, dinv, b1, xb, NH, 1);

    // --- layer 2: h2 = relu(h1agg) @ W2^T ---
    k_f32_to_bf16<<<blks((long long)H * H, 256), 256, 0, stream>>>(W2, wb, (long long)H * H);
    {
        dim3 g(blks(N, 128), H / 64);
        k_gemm_bf16<4><<<g, 256, 0, stream>>>(xb, wb, hbuf, nullptr, N, H, H, H);
    }
    k_zero_f32<<<blks(NH, 256), 256, 0, stream>>>(aggbuf, NH);
    k_aggregate<<<blks(E, 8), 256, 0, stream>>>(src, dst, dinv, hbuf, aggbuf, E);
    k_combine<<<blks(NH, 256), 256, 0, stream>>>(aggbuf, hbuf, dinv, b2, xb, NH, 0);

    // --- output layer: logits = h2agg @ Wout^T + bout ---
    k_conv_pad<<<blks((long long)CPAD * H, 256), 256, 0, stream>>>(Wout, wb, C, CPAD, H);
    {
        dim3 g(blks(N, 128), 1);
        k_gemm_bf16<3><<<g, 256, 0, stream>>>(xb, wb, (float*)d_out, bout, N, H, C, C);
    }
}
